// MultiHeadSelfAttention_50912542327533
// MI455X (gfx1250) — compile-verified, matrix operand order checked by reading, not run on hardware
//
#include <hip/hip_runtime.h>
#include <hip/hip_bf16.h>

// ---------------------------------------------------------------------------
// MI455X (gfx1250) multi-head self-attention forward, wave32 WMMA bf16.
//   K0: one-shot fp32 -> bf16 conversion of x, Wqkv, Wout
//   K1: qkv = x @ Wqkv^T     (bf16 WMMA, 64x64 tile/wave: 1 b128 load / WMMA)
//   K2: RoPE + pack Q,K [B,H,S,dk] and V^T [B,H,dk,S] in bf16
//   K3: flash attention, online softmax, P transposed via LDS
//   K4: out = O @ Wout^T     (same 64x64 GEMM) -> d_out fp32
// ---------------------------------------------------------------------------

typedef __attribute__((ext_vector_type(16))) __bf16 v16bf;
typedef __attribute__((ext_vector_type(8)))  __bf16 v8bf;
typedef __attribute__((ext_vector_type(4)))  __bf16 v4bf;
typedef __attribute__((ext_vector_type(8)))  float  v8f;

#define D_MODEL 1024
#define NHEADS  16
#define DK      64
#define SEQ     2048

static __device__ inline v8f wmma_bf16(v16bf a, v16bf b, v8f c) {
  return __builtin_amdgcn_wmma_f32_16x16x32_bf16(false, a, false, b,
                                                 (short)0, c, false, false);
}

// A/B fragment of a 16x32 bf16 tile; operand row contiguous in memory.
// kbase must already include +8*(lane>=16). Halves 0..7 = k+0..7, 8..15 = k+16..23.
static __device__ inline v16bf frag_from_bf16(const __bf16* __restrict__ row, int kbase) {
  v8bf lo = *reinterpret_cast<const v8bf*>(row + kbase);
  v8bf hi = *reinterpret_cast<const v8bf*>(row + kbase + 16);
  v16bf r;
#pragma unroll
  for (int i = 0; i < 8; ++i) { r[i] = lo[i]; r[8 + i] = hi[i]; }
  return r;
}

// ---------------------------------------------------------------------------
// K0: fp32 -> bf16, 4 elements/thread.
// ---------------------------------------------------------------------------
__global__ void __launch_bounds__(256)
cvt_bf16_kernel(const float* __restrict__ src, __bf16* __restrict__ dst, int n4) {
  const int i = blockIdx.x * blockDim.x + threadIdx.x;
  if (i < n4) {
    const float4 v = reinterpret_cast<const float4*>(src)[i];
    v4bf o;
    o[0] = (__bf16)v.x; o[1] = (__bf16)v.y; o[2] = (__bf16)v.z; o[3] = (__bf16)v.w;
    reinterpret_cast<v4bf*>(dst)[i] = o;
  }
}

// ---------------------------------------------------------------------------
// K1/K4: C[M, N] = A[M,1024](bf16) @ B[N,1024]^T(bf16) -> fp32, ldc = N.
// 64x64 tile per wave: per k-step 4 A-frags + 4 B-frags -> 16 WMMAs
// (1 global_load_b128 per WMMA). ~185 VGPRs, no LDS needed: all operands
// are L2-resident (192 MB) and row pointers fold into 24-bit imm offsets.
// ---------------------------------------------------------------------------
__global__ void __launch_bounds__(128)
gemm64_kernel(const __bf16* __restrict__ A, const __bf16* __restrict__ Bm,
              float* __restrict__ C, int ldc) {
  const int lane = threadIdx.x & 31;
  const int wave = threadIdx.x >> 5;
  const int ln = lane & 15, hi = lane >> 4;
  const int nt = blockIdx.x * 2 + (wave & 1);   // 64-col tile index
  const int mt = blockIdx.y * 2 + (wave >> 1);  // 64-row tile index

  const __bf16* abase = A  + (size_t)(mt * 64 + ln) * D_MODEL;
  const __bf16* bbase = Bm + (size_t)(nt * 64 + ln) * D_MODEL;

  v8f acc[4][4];
#pragma unroll
  for (int mi = 0; mi < 4; ++mi)
#pragma unroll
    for (int ni = 0; ni < 4; ++ni) acc[mi][ni] = v8f{};

#pragma unroll 2
  for (int kk = 0; kk < D_MODEL; kk += 32) {
    __builtin_prefetch(abase + kk + 256, 0, 1);
    __builtin_prefetch(bbase + kk + 256, 0, 1);
    const int kb = kk + hi * 8;
    v16bf a[4];
#pragma unroll
    for (int mi = 0; mi < 4; ++mi)
      a[mi] = frag_from_bf16(abase + (size_t)mi * 16 * D_MODEL, kb);
#pragma unroll
    for (int ni = 0; ni < 4; ++ni) {
      const v16bf b = frag_from_bf16(bbase + (size_t)ni * 16 * D_MODEL, kb);
#pragma unroll
      for (int mi = 0; mi < 4; ++mi)
        acc[mi][ni] = wmma_bf16(a[mi], b, acc[mi][ni]);
    }
  }

  float* crow = C + (size_t)(mt * 64 + hi * 8) * ldc + nt * 64 + ln;
#pragma unroll
  for (int mi = 0; mi < 4; ++mi)
#pragma unroll
    for (int r = 0; r < 8; ++r)
#pragma unroll
      for (int ni = 0; ni < 4; ++ni)
        crow[(size_t)(mi * 16 + r) * ldc + ni * 16] = acc[mi][ni][r];
}

// ---------------------------------------------------------------------------
// K2: RoPE on q,k + pack to bf16. One thread per (b,h,s,pair).
// ---------------------------------------------------------------------------
__global__ void __launch_bounds__(256)
rope_pack_kernel(const float* __restrict__ qkv, const int* __restrict__ pos,
                 __bf16* __restrict__ Qh, __bf16* __restrict__ Kh,
                 __bf16* __restrict__ Vt) {
  const int idx = blockIdx.x * blockDim.x + threadIdx.x;   // 2^21 threads
  const int i = idx & 31;
  const int s = (idx >> 5) & (SEQ - 1);
  const int h = (idx >> 16) & (NHEADS - 1);
  const int b = (idx >> 20) & 1;

  const int d0 = 2 * i, d1 = 2 * i + 1;
  const size_t row = ((size_t)b * SEQ + s) * 3072;
  const int col = h * DK;

  const float q0 = qkv[row + col + d0],        q1 = qkv[row + col + d1];
  const float k0 = qkv[row + 1024 + col + d0], k1 = qkv[row + 1024 + col + d1];
  const float v0 = qkv[row + 2048 + col + d0], v1 = qkv[row + 2048 + col + d1];

  const float p = (float)pos[s];
  const float fr = __expf(-((float)d0) * (9.210340371976184f / 64.0f)); // theta^-d0/64
  const float ang = p * fr;
  const float cs = __cosf(ang), sn = __sinf(ang);

  const int bh = b * NHEADS + h;
  const size_t qrow = ((size_t)bh * SEQ + s) * DK;
  Qh[qrow + d0] = (__bf16)(q0 * cs - q1 * sn);
  Qh[qrow + d1] = (__bf16)(q0 * sn + q1 * cs);
  Kh[qrow + d0] = (__bf16)(k0 * cs - k1 * sn);
  Kh[qrow + d1] = (__bf16)(k0 * sn + k1 * cs);

  const size_t vb = (size_t)bh * DK;
  Vt[(vb + d0) * SEQ + s] = (__bf16)v0;
  Vt[(vb + d1) * SEQ + s] = (__bf16)v1;
}

// 16-lane reductions (xor masks <=8 stay within each wave32 half).
static __device__ inline float rmax16(float v) {
#pragma unroll
  for (int m = 8; m >= 1; m >>= 1) v = fmaxf(v, __shfl_xor(v, m));
  return v;
}
static __device__ inline float rsum16(float v) {
#pragma unroll
  for (int m = 8; m >= 1; m >>= 1) v += __shfl_xor(v, m);
  return v;
}

// ---------------------------------------------------------------------------
// K3: flash attention. One wave = 16 queries x full key sequence, 32 keys/iter.
// ---------------------------------------------------------------------------
__global__ void __launch_bounds__(128)
attn_kernel(const __bf16* __restrict__ Qh, const __bf16* __restrict__ Kh,
            const __bf16* __restrict__ Vt, __bf16* __restrict__ O) {
  __shared__ __attribute__((aligned(16))) __bf16 lds[4][16 * 32];

  const int lane = threadIdx.x & 31;
  const int wave = threadIdx.x >> 5;
  const int ln = lane & 15, hi = lane >> 4;

  const int tile = blockIdx.x * 4 + wave;      // 0..4095
  const int qt = tile & 127;
  const int bh = tile >> 7;
  const int h = bh & (NHEADS - 1), b = bh >> 4;
  const int q0 = qt * 16;

  const __bf16* qrow = Qh + ((size_t)bh * SEQ + q0 + ln) * DK;
  const v16bf qa0 = frag_from_bf16(qrow, 0 + hi * 8);
  const v16bf qa1 = frag_from_bf16(qrow, 32 + hi * 8);

  v8f acc[4];
#pragma unroll
  for (int n = 0; n < 4; ++n) acc[n] = v8f{};
  float m_run[8], l_run[8];
#pragma unroll
  for (int r = 0; r < 8; ++r) { m_run[r] = -INFINITY; l_run[r] = 0.0f; }

  __bf16* pl = &lds[wave][0];
  const __bf16* kbase0 = Kh + (size_t)bh * SEQ * DK;
  const __bf16* vbase0 = Vt + (size_t)bh * DK * SEQ;

  for (int k0 = 0; k0 < SEQ; k0 += 32) {
    v8f c0 = {}, c1 = {};
    {
      const __bf16* krow0 = kbase0 + (size_t)(k0 + ln) * DK;
      const __bf16* krow1 = kbase0 + (size_t)(k0 + 16 + ln) * DK;
      c0 = wmma_bf16(qa0, frag_from_bf16(krow0, 0 + hi * 8), c0);
      c0 = wmma_bf16(qa1, frag_from_bf16(krow0, 32 + hi * 8), c0);
      c1 = wmma_bf16(qa0, frag_from_bf16(krow1, 0 + hi * 8), c1);
      c1 = wmma_bf16(qa1, frag_from_bf16(krow1, 32 + hi * 8), c1);
    }

    // online softmax, scale = 1/sqrt(64)
#pragma unroll
    for (int r = 0; r < 8; ++r) {
      const float s0 = c0[r] * 0.125f;
      const float s1 = c1[r] * 0.125f;
      const float tmax = rmax16(fmaxf(s0, s1));
      const float mn = fmaxf(m_run[r], tmax);
      const float alpha = __expf(m_run[r] - mn);
      const float p0 = __expf(s0 - mn);
      const float p1 = __expf(s1 - mn);
      l_run[r] = l_run[r] * alpha + rsum16(p0 + p1);
      m_run[r] = mn;
#pragma unroll
      for (int n = 0; n < 4; ++n) acc[n][r] *= alpha;
      const int m = r + hi * 8;                // C-layout row
      pl[m * 32 + ln]      = (__bf16)p0;
      pl[m * 32 + 16 + ln] = (__bf16)p1;
    }
    asm volatile("s_wait_dscnt 0" ::: "memory");

    const v16bf pa = frag_from_bf16(pl + ln * 32, hi * 8);
#pragma unroll
    for (int n = 0; n < 4; ++n) {
      const __bf16* vrow = vbase0 + (size_t)(n * 16 + ln) * SEQ + k0;
      acc[n] = wmma_bf16(pa, frag_from_bf16(vrow, hi * 8), acc[n]);
    }
  }

  __bf16* obase = O + ((size_t)b * SEQ + q0 + hi * 8) * D_MODEL + h * DK + ln;
#pragma unroll
  for (int r = 0; r < 8; ++r) {
    const float inv = 1.0f / l_run[r];
#pragma unroll
    for (int n = 0; n < 4; ++n)
      obase[(size_t)r * D_MODEL + n * 16] = (__bf16)(acc[n][r] * inv);
  }
}

// ---------------------------------------------------------------------------
extern "C" void kernel_launch(void* const* d_in, const int* in_sizes, int n_in,
                              void* d_out, int out_size, void* d_ws, size_t ws_size,
                              hipStream_t stream) {
  (void)in_sizes; (void)n_in; (void)out_size; (void)ws_size;

  const float* x    = (const float*)d_in[0];   // [2,2048,1024]
  const float* Wqkv = (const float*)d_in[1];   // [3072,1024]
  const float* Wout = (const float*)d_in[2];   // [1024,1024]
  const int*   pos  = (const int*)d_in[3];     // [2048]
  float* out = (float*)d_out;                  // [2,2048,1024]

  char* ws = (char*)d_ws;
  float*  qkv   = (float*)(ws);                 // 50,331,648 B
  __bf16* Qh    = (__bf16*)(ws + 50331648);     //  8,388,608 B
  __bf16* Kh    = (__bf16*)(ws + 58720256);     //  8,388,608 B
  __bf16* Vt    = (__bf16*)(ws + 67108864);     //  8,388,608 B
  __bf16* O     = (__bf16*)(ws + 75497472);     //  8,388,608 B
  __bf16* xb    = (__bf16*)(ws + 83886080);     //  8,388,608 B
  __bf16* Wqkvb = (__bf16*)(ws + 92274688);     //  6,291,456 B
  __bf16* Woutb = (__bf16*)(ws + 98566144);     //  2,097,152 B

  // K0: one-shot downconversions (memory bound, ~34 MB total)
  cvt_bf16_kernel<<<4096, 256, 0, stream>>>(x,    xb,    1048576);
  cvt_bf16_kernel<<<3072, 256, 0, stream>>>(Wqkv, Wqkvb, 786432);
  cvt_bf16_kernel<<<1024, 256, 0, stream>>>(Wout, Woutb, 262144);

  // K1: M=4096 (64 m64-tiles), N=3072 (48 n64-tiles)
  dim3 g1(24, 32);
  gemm64_kernel<<<g1, 128, 0, stream>>>(xb, Wqkvb, qkv, 3072);

  rope_pack_kernel<<<8192, 256, 0, stream>>>(qkv, pos, Qh, Kh, Vt);

  attn_kernel<<<1024, 128, 0, stream>>>(Qh, Kh, Vt, O);

  // K4: M=4096 (64 m64-tiles), N=1024 (16 n64-tiles)
  dim3 g2(8, 32);
  gemm64_kernel<<<g2, 128, 0, stream>>>(O, Woutb, out, 1024);
}
